// MSCA_52321291600536
// MI455X (gfx1250) — matrix operand order repaired and checked by reading, not compiled, not run on hardware
//
#include <hip/hip_runtime.h>

// ---------------------------------------------------------------------------
// MSCA block for MI455X (gfx1250, wave32, WMMA).
// Attention-out + projection fused into bf16 WMMA GEMMs:
//   final[b,n,:] = proj_b + sum_h softmax(scale*q[b,h,n]*k[b,h,:]) @ W2[b,h]
//   W2[b,h][m,co] = sum_d v[b,h,m,d] * proj_w[co, h*8+d]
// ---------------------------------------------------------------------------

#define BB    8
#define HH    128
#define CC    64
#define HEADS 8
#define HD    8
#define NN    (HH*HH)   // 16384 tokens
#define NP    64        // 8x8 reduced tokens
#define SCALE 0.35355339059327373f   // 8^-0.5

typedef __attribute__((ext_vector_type(8)))  float  v8f;
typedef __attribute__((ext_vector_type(16))) __bf16 v16bf;
typedef int i32x4g __attribute__((vector_size(16)));   // matches builtin's V4i

#if defined(__has_builtin)
#if __has_builtin(__builtin_amdgcn_global_load_async_to_lds_b128) && \
    __has_builtin(__builtin_amdgcn_s_wait_asynccnt)
#define HAVE_ASYNC_LDS 1
#endif
#endif
#ifndef HAVE_ASYNC_LDS
#define HAVE_ASYNC_LDS 0
#endif

union BfVec {            // 32B: one WMMA 16-bit operand (8 VGPRs)
  v16bf  v;
  float4 f[2];
};

__device__ __forceinline__ __bf16 f2bf(float x) {
  unsigned u = __builtin_bit_cast(unsigned, x);
  unsigned r = u + 0x7FFFu + ((u >> 16) & 1u);       // round-to-nearest-even
  unsigned short hs = (unsigned short)(r >> 16);
  return __builtin_bit_cast(__bf16, hs);
}

// ---------------------------------------------------------------------------
// q[b,h,n] = sum_c x[b,n,c] * q_w[h,c]
// ---------------------------------------------------------------------------
__global__ void qproj_kernel(const float* __restrict__ x,
                             const float* __restrict__ q_w,
                             float* __restrict__ q) {
  __shared__ float sqw[HEADS*CC];
  int tid = threadIdx.x;
  for (int i = tid; i < HEADS*CC; i += 256) sqw[i] = q_w[i];
  __syncthreads();
  int gid = blockIdx.x * 256 + tid;          // exactly B*N threads
  int b = gid / NN, n = gid % NN;
  const float4* xr = (const float4*)(x + ((size_t)b*NN + n)*CC);
  float acc[HEADS];
#pragma unroll
  for (int h = 0; h < HEADS; ++h) acc[h] = 0.f;
#pragma unroll
  for (int i = 0; i < CC/4; ++i) {
    float4 xv = xr[i];
#pragma unroll
    for (int h = 0; h < HEADS; ++h) {
      const float* w = &sqw[h*CC + i*4];
      acc[h] += xv.x*w[0] + xv.y*w[1] + xv.z*w[2] + xv.w*w[3];
    }
  }
#pragma unroll
  for (int h = 0; h < HEADS; ++h)
    q[(size_t)(b*HEADS + h)*NN + n] = acc[h];
}

// ---------------------------------------------------------------------------
// stride-2 depthwise 2x2 conv, channel-last [b][y][x][c]
// ---------------------------------------------------------------------------
__global__ void redconv_kernel(const float* __restrict__ in, float* __restrict__ out,
                               const float* __restrict__ w, const float* __restrict__ bias,
                               int Hin, int Hout) {
  int gid = blockIdx.x*256 + threadIdx.x;
  int total = BB*Hout*Hout*CC;
  if (gid >= total) return;
  int c = gid % CC;
  int s = gid / CC;
  int xo = s % Hout; int yo = (s / Hout) % Hout; int b = s / (Hout*Hout);
  const float* wr = w + c*4;                       // red_w[c,0,dy,dx]
  const float* base = in + (((size_t)b*Hin + 2*yo)*Hin + 2*xo)*CC + c;
  float acc = bias[c];
  acc += base[0]                   * wr[0];
  acc += base[CC]                  * wr[1];
  acc += base[(size_t)Hin*CC]      * wr[2];
  acc += base[(size_t)Hin*CC + CC] * wr[3];
  out[gid] = acc;
}

// depthwise 3x3, pad 1, on 8x8 map, channel-last
__global__ void dwconv_kernel(const float* __restrict__ in, float* __restrict__ out,
                              const float* __restrict__ w, const float* __restrict__ bias) {
  int gid = blockIdx.x*256 + threadIdx.x;          // exactly B*8*8*64
  int c = gid % CC; int s = gid / CC;
  int xo = s % 8; int yo = (s/8) % 8; int b = s / 64;
  const float* wr = w + c*9;
  float acc = bias[c];
#pragma unroll
  for (int dy = 0; dy < 3; ++dy) {
    int yy = yo + dy - 1; if (yy < 0 || yy > 7) continue;
#pragma unroll
    for (int dx = 0; dx < 3; ++dx) {
      int xx = xo + dx - 1; if (xx < 0 || xx > 7) continue;
      acc += in[(((size_t)b*8 + yy)*8 + xx)*CC + c] * wr[dy*3+dx];
    }
  }
  out[gid] = acc;
}

// ---------------------------------------------------------------------------
// Per-batch: 1x1 conv + bias, LN, exact GELU, then k, v, and folded
// W2T[b][h][co][m] = sum_d v[m][h*8+d]*proj_w[co][h*8+d]  (bf16, transposed
// so attention-kernel B operands are contiguous float4 LDS loads)
// ---------------------------------------------------------------------------
__global__ void fuse_kernel(const float* __restrict__ t5,
                            const float* __restrict__ conv_w, const float* __restrict__ conv_b,
                            const float* __restrict__ ln_w,   const float* __restrict__ ln_b,
                            const float* __restrict__ k_w,    const float* __restrict__ v_w,
                            const float* __restrict__ proj_w,
                            float* __restrict__ kbuf, __bf16* __restrict__ W2T) {
  __shared__ float sX[NP*CC];
  __shared__ float sY[NP*CC];
  __shared__ float sV[NP*CC];
  int b = blockIdx.x, tid = threadIdx.x;
  for (int i = tid; i < NP*CC; i += 256) sX[i] = t5[(size_t)b*NP*CC + i];
  __syncthreads();
  // pointwise conv C->C
  for (int o = tid; o < NP*CC; o += 256) {
    int c = o % CC, m = o / CC;
    float acc = conv_b[c];
    const float* wr = conv_w + c*CC;
    const float* xr = sX + m*CC;
#pragma unroll 8
    for (int ci = 0; ci < CC; ++ci) acc += xr[ci]*wr[ci];
    sY[o] = acc;
  }
  __syncthreads();
  // LayerNorm + exact GELU, one thread per token
  if (tid < NP) {
    float* r = sY + tid*CC;
    float mu = 0.f;
    for (int c = 0; c < CC; ++c) mu += r[c];
    mu *= (1.f/CC);
    float var = 0.f;
    for (int c = 0; c < CC; ++c) { float d = r[c]-mu; var += d*d; }
    var *= (1.f/CC);
    float inv = rsqrtf(var + 1e-5f);
    for (int c = 0; c < CC; ++c) {
      float v = (r[c]-mu)*inv*ln_w[c] + ln_b[c];
      r[c] = 0.5f*v*(1.f + erff(v*0.70710678118f));
    }
  }
  __syncthreads();
  // k[b,h,m]
  for (int o = tid; o < HEADS*NP; o += 256) {
    int h = o / NP, m = o % NP;
    float acc = 0.f;
    const float* wr = k_w + h*CC;
    const float* yr = sY + m*CC;
#pragma unroll 8
    for (int c = 0; c < CC; ++c) acc += yr[c]*wr[c];
    kbuf[(b*HEADS + h)*NP + m] = acc;
  }
  // v[m][h*8+d]
  for (int o = tid; o < NP*CC; o += 256) {
    int dv = o % CC, m = o / CC;
    float acc = 0.f;
    const float* wr = v_w + dv*CC;
    const float* yr = sY + m*CC;
#pragma unroll 8
    for (int c = 0; c < CC; ++c) acc += yr[c]*wr[c];
    sV[m*CC + dv] = acc;
  }
  __syncthreads();
  // W2T (transposed, bf16)
  for (int o = tid; o < HEADS*CC*NP; o += 256) {
    int h  = o / (CC*NP);
    int co = (o / NP) % CC;
    int m  = o % NP;
    float acc = 0.f;
    const float* pw = proj_w + co*CC + h*HD;
    const float* vr = sV + m*CC + h*HD;
#pragma unroll
    for (int d = 0; d < HD; ++d) acc += vr[d]*pw[d];
    W2T[((size_t)(b*HEADS + h)*CC + co)*NP + m] = f2bf(acc);
  }
}

// ---------------------------------------------------------------------------
// Attention + projection via v_wmma_f32_16x16x32_bf16.
// One wave = 16 tokens, full 64 output channels (4 accumulators).
// Block = 8 waves, all same b; W2[b,h] staged into LDS (async DMA when the
// toolchain exposes GLOBAL_LOAD_ASYNC_TO_LDS), overlapped with softmax math.
// ---------------------------------------------------------------------------
__global__ void __launch_bounds__(256)
attn_kernel(const float* __restrict__ q, const float* __restrict__ kbuf,
            const __bf16* __restrict__ W2T, const float* __restrict__ proj_b,
            float* __restrict__ out) {
  __shared__ __bf16 sW2[CC*NP];       // current (b,h): [co][m], 8KB
  __shared__ __bf16 sA[8*16*NP];      // per-wave softmax weights [t][m], 16KB
  int tid  = threadIdx.x;
  int wave = tid >> 5, lane = tid & 31;
  int tile = blockIdx.x*8 + wave;     // 8192 tiles = B * N/16
  int b  = tile >> 10;
  int n0 = (tile & 1023) << 4;
  int t    = lane & 15;               // token row in tile / B-column
  int half = lane >> 4;

  v8f acc[4];
#pragma unroll
  for (int j = 0; j < 4; ++j) {
    float pv = proj_b[j*16 + t];
#pragma unroll
    for (int i = 0; i < 8; ++i) acc[j][i] = pv;
  }

  for (int h = 0; h < HEADS; ++h) {
    __syncthreads();   // all waves finished reading previous sW2/sA

    // ---- stage W2T[b][h] (8KB bf16) into LDS, overlapped with softmax ----
    const char* gsrc = (const char*)(W2T + (size_t)(b*HEADS + h)*CC*NP);
#if HAVE_ASYNC_LDS
    {
      char* ldst = (char*)sW2;
      __builtin_amdgcn_global_load_async_to_lds_b128(
          (__attribute__((address_space(1))) i32x4g*)(gsrc + (size_t)tid*16),
          (__attribute__((address_space(3))) i32x4g*)(ldst + (size_t)tid*16), 0, 0);
      __builtin_amdgcn_global_load_async_to_lds_b128(
          (__attribute__((address_space(1))) i32x4g*)(gsrc + 4096 + (size_t)tid*16),
          (__attribute__((address_space(3))) i32x4g*)(ldst + 4096 + (size_t)tid*16), 0, 0);
    }
#else
    {
      const unsigned int* src = (const unsigned int*)gsrc;
      unsigned int* dst = (unsigned int*)sW2;
      for (int i = tid; i < CC*NP/2; i += 256) dst[i] = src[i];
    }
#endif
    // prefetch next head's W2 into cache (global_prefetch_b8)
    if (h + 1 < HEADS)
      __builtin_prefetch(gsrc + (size_t)CC*NP*2 + (size_t)tid*32, 0, 1);

    // ---- rank-1 logits + softmax over m (lane pair covers 32 m's each) ----
    float qv = q[(size_t)(b*HEADS + h)*NN + n0 + t] * SCALE;
    const float4* k4 = (const float4*)(kbuf + (b*HEADS + h)*NP + half*32);
    float e[32];
    float mymax = -3.4e38f;
#pragma unroll
    for (int i = 0; i < 8; ++i) {
      float4 kv = k4[i];
      e[4*i+0] = qv*kv.x; e[4*i+1] = qv*kv.y;
      e[4*i+2] = qv*kv.z; e[4*i+3] = qv*kv.w;
      mymax = fmaxf(mymax, fmaxf(fmaxf(e[4*i+0], e[4*i+1]),
                                 fmaxf(e[4*i+2], e[4*i+3])));
    }
    float rmax = fmaxf(mymax, __shfl_xor(mymax, 16));
    float sum = 0.f;
#pragma unroll
    for (int i = 0; i < 32; ++i) { float ev = __expf(e[i]-rmax); e[i] = ev; sum += ev; }
    float tot = sum + __shfl_xor(sum, 16);
    float inv = 1.f / tot;
    __bf16* wrow = sA + wave*16*NP + t*NP + half*32;
#pragma unroll
    for (int i = 0; i < 32; ++i) wrow[i] = f2bf(e[i]*inv);

#if HAVE_ASYNC_LDS
    __builtin_amdgcn_s_wait_asynccnt(0);
#endif
    __syncthreads();   // staged W2 + all sA rows visible

    // A operand (16-bit A 16x32 layout): lanes 0-15 row t, elems = K{0..7,16..23};
    // lanes 16-31 row t, elems = K{8..15,24..31}. Each = two contiguous float4s.
    const float4* arow = (const float4*)(sA + wave*16*NP + t*NP);
    BfVec a0, a1;
    a0.f[0] = arow[half];     a0.f[1] = arow[half+2];   // K 0..31 chunk
    a1.f[0] = arow[half+4];   a1.f[1] = arow[half+6];   // K 32..63 chunk

    // Preload ALL B operands, then issue 8 WMMAs back-to-back (no per-tile
    // ds_load -> wait -> wmma serialization).
    BfVec b0[4], b1[4];
#pragma unroll
    for (int j = 0; j < 4; ++j) {
      // B operand (16-bit B 32x16): lanes 0-15 K{0..7,16..23}, lanes 16-31 K{8..15,24..31},
      // column = lane&15. W2T row [co][m] is contiguous in m=K.
      const float4* brow = (const float4*)(sW2 + (j*16 + t)*NP);
      b0[j].f[0] = brow[half];     b0[j].f[1] = brow[half+2];
      b1[j].f[0] = brow[half+4];   b1[j].f[1] = brow[half+6];
    }
#pragma unroll
    for (int j = 0; j < 4; ++j) {
      acc[j] = __builtin_amdgcn_wmma_f32_16x16x32_bf16(
          false, a0.v, false, b0[j].v, (short)0, acc[j], false, false);
      acc[j] = __builtin_amdgcn_wmma_f32_16x16x32_bf16(
          false, a1.v, false, b1[j].v, (short)0, acc[j], false, false);
    }
  }

  // D layout: VGPR r -> (M=r, N=lane) for lanes 0-15, (M=8+r, N=lane-16) else
  float* orow = out + ((size_t)b*NN + n0)*CC;
#pragma unroll
  for (int j = 0; j < 4; ++j)
#pragma unroll
    for (int r = 0; r < 8; ++r)
      orow[(half*8 + r)*CC + j*16 + t] = acc[j][r];
}

// ---------------------------------------------------------------------------
extern "C" void kernel_launch(void* const* d_in, const int* in_sizes, int n_in,
                              void* d_out, int out_size, void* d_ws, size_t ws_size,
                              hipStream_t stream) {
  const float* x      = (const float*)d_in[0];
  const float* q_w    = (const float*)d_in[1];
  const float* k_w    = (const float*)d_in[2];
  const float* v_w    = (const float*)d_in[3];
  const float* proj_w = (const float*)d_in[4];
  const float* proj_b = (const float*)d_in[5];
  const float* red_w  = (const float*)d_in[6];
  const float* red_b  = (const float*)d_in[7];
  const float* dw_w   = (const float*)d_in[8];
  const float* dw_b   = (const float*)d_in[9];
  const float* conv_w = (const float*)d_in[10];
  const float* conv_b = (const float*)d_in[11];
  const float* ln_w   = (const float*)d_in[12];
  const float* ln_b   = (const float*)d_in[13];
  float* out = (float*)d_out;

  char* ws = (char*)d_ws;
  size_t off = 0;
  auto alloc = [&](size_t bytes) {
    char* p = ws + off;
    off += (bytes + 255) & ~(size_t)255;
    return p;
  };
  float*  qbuf = (float*) alloc((size_t)BB*HEADS*NN*sizeof(float));   // 4 MB
  float*  t1   = (float*) alloc((size_t)BB*64*64*CC*sizeof(float));   // 8 MB
  float*  t2   = (float*) alloc((size_t)BB*32*32*CC*sizeof(float));
  float*  t3   = (float*) alloc((size_t)BB*16*16*CC*sizeof(float));
  float*  t4   = (float*) alloc((size_t)BB*8*8*CC*sizeof(float));
  float*  t5   = (float*) alloc((size_t)BB*8*8*CC*sizeof(float));
  float*  kbuf = (float*) alloc((size_t)BB*HEADS*NP*sizeof(float));
  __bf16* W2T  = (__bf16*)alloc((size_t)BB*HEADS*CC*NP*sizeof(__bf16));

  qproj_kernel<<<BB*NN/256, 256, 0, stream>>>(x, q_w, qbuf);
  redconv_kernel<<<(BB*64*64*CC)/256, 256, 0, stream>>>(x,  t1, red_w, red_b, 128, 64);
  redconv_kernel<<<(BB*32*32*CC)/256, 256, 0, stream>>>(t1, t2, red_w, red_b,  64, 32);
  redconv_kernel<<<(BB*16*16*CC)/256, 256, 0, stream>>>(t2, t3, red_w, red_b,  32, 16);
  redconv_kernel<<<(BB*8*8*CC)/256,   256, 0, stream>>>(t3, t4, red_w, red_b,  16,  8);
  dwconv_kernel<<<(BB*8*8*CC)/256, 256, 0, stream>>>(t4, t5, dw_w, dw_b);
  fuse_kernel<<<BB, 256, 0, stream>>>(t5, conv_w, conv_b, ln_w, ln_b,
                                      k_w, v_w, proj_w, kbuf, W2T);
  attn_kernel<<<(BB*NN/16)/8, 256, 0, stream>>>(qbuf, kbuf, W2T, proj_b, out);
}
